// RetinaNet_18691697672674
// MI455X (gfx1250) — matrix operand order repaired and checked by reading, not compiled, not run on hardware
//
#include <hip/hip_runtime.h>

typedef _Float16 f16;
typedef _Float16 v16h __attribute__((ext_vector_type(16)));
typedef _Float16 v8h  __attribute__((ext_vector_type(8)));
typedef float    v8f  __attribute__((ext_vector_type(8)));

#define CIN     256
#define KTOT    2304   // 9 * 256
#define KCHUNKS 72     // 2304 / 32
#define CLS_STRIDE 768 // padded 720
#define REG_STRIDE 64  // padded 36

// ---------------------------------------------------------------------------
// fp32 NCHW -> fp16 NHWC feature conversion: out[pos*256 + c] = in[c*HW + pos]
// ---------------------------------------------------------------------------
__global__ void cvt_nchw_to_nhwc_f16(const float* __restrict__ in, f16* __restrict__ out, int HW) {
    int i = blockIdx.x * blockDim.x + threadIdx.x;
    if (i >= HW * CIN) return;
    int pos = i >> 8;
    int c = i & 255;
    out[i] = (f16)in[c * HW + pos];
}

// ---------------------------------------------------------------------------
// Repack conv weights [O][256][3][3] f32 -> [Opad][K] f16 with K = tap*256 + c
// (rows o >= O zero-padded so M groups are full 64-row groups)
// ---------------------------------------------------------------------------
__global__ void wt_transform(const float* __restrict__ w, f16* __restrict__ wt, int O, int Opad) {
    int i = blockIdx.x * blockDim.x + threadIdx.x;
    int n = Opad * KTOT;
    if (i >= n) return;
    int o = i / KTOT;
    int k = i - o * KTOT;
    int t = k >> 8;       // tap 0..8
    int c = k & 255;      // channel
    float v = (o < O) ? w[(o * CIN + c) * 9 + t] : 0.f;
    wt[i] = (f16)v;
}

// ---------------------------------------------------------------------------
// Implicit-GEMM 3x3 SAME conv, 256 in-channels, NHWC activations, via
// v_wmma_f32_16x16x32_f16. One wave = 64(M=oc) x 16(N=pos) tile:
// 4 register-blocked accumulators sharing one B fragment per 32-wide K chunk.
// A/B fragments double-buffered so next-chunk loads overlap current WMMAs.
//
// A lane layout (16-bit A 16x32): lanes 0-15 K{0-7,16-23}, lanes 16-31 K{8-15,24-31}
// B lane layout (16-bit B 32x16): lane&15 = column, same K split as A (lanes>=16 hold K+8)
// C/D layout: lane&15 = column N, VGPR r = row r + 8*(lane>=16)
// ---------------------------------------------------------------------------
__global__ __launch_bounds__(128)
void conv3x3_wmma(const f16* __restrict__ in,     // [HW][256] f16 NHWC
                  const f16* __restrict__ wt,     // [Opad][2304] f16, tap-major K
                  const float* __restrict__ bias, // [O]
                  f16* __restrict__ out16,        // [HW][256] f16 (inner layers, O==256) or null
                  float* __restrict__ out32,      // [HW][Opad] f32 (heads) or null
                  int H, int W, int HW, int O, int Opad, int nTilesN, int T, int relu)
{
    int tile = blockIdx.x * 4 + (threadIdx.x >> 5);
    if (tile >= T) return;                 // wave-uniform exit (EXEC stays all-1)
    int lane = threadIdx.x & 31;
    int mg = tile / nTilesN;               // 64-output-channel group
    int nt = tile - mg * nTilesN;

    int lane15 = lane & 15;
    int hi = lane >> 4;                    // 0 or 1
    int aoff = hi ? 8 : 0;                 // A K sub-offset for hi lanes
    int koff = hi ? 8 : 0;                 // B K (=channel) sub-offset for hi lanes

    int n = nt * 16 + lane15;              // this lane's output position (B column)
    bool nvalid = (n < HW);
    int nn = nvalid ? n : 0;
    int y = nn / W;
    int x = nn - y * W;

    const f16* arow0 = wt + (size_t)(mg * 64 +  0 + lane15) * KTOT + aoff;
    const f16* arow1 = wt + (size_t)(mg * 64 + 16 + lane15) * KTOT + aoff;
    const f16* arow2 = wt + (size_t)(mg * 64 + 32 + lane15) * KTOT + aoff;
    const f16* arow3 = wt + (size_t)(mg * 64 + 48 + lane15) * KTOT + aoff;

    auto loadArow = [&](const f16* row, int kbase, v16h& A) {
        v8h lo = *(const v8h*)(row + kbase);
        v8h hh = *(const v8h*)(row + kbase + 16);
        #pragma unroll
        for (int i = 0; i < 8; ++i) { A[i] = lo[i]; A[i + 8] = hh[i]; }
    };
    auto loadB = [&](int kc, v16h& B) {
        int kbase = kc * 32;
        int tap = kc >> 3;                 // 8 chunks per tap
        int ty = tap / 3;
        int tx = tap - ty * 3;
        int iy = y + ty - 1;
        int ix = x + tx - 1;
        v16h z = {};
        B = z;
        if (nvalid && iy >= 0 && iy < H && ix >= 0 && ix < W) {
            // NHWC: 32 chunk channels are contiguous at this pixel
            const f16* bp = in + (size_t)(iy * W + ix) * CIN + (kbase & 255) + koff;
            v8h lo = *(const v8h*)bp;
            v8h hh = *(const v8h*)(bp + 16);
            #pragma unroll
            for (int i = 0; i < 8; ++i) { B[i] = lo[i]; B[i + 8] = hh[i]; }
        }
    };

    v8f acc0 = {}, acc1 = {}, acc2 = {}, acc3 = {};
    v16h A0, A1, A2, A3, B;
    v16h nA0, nA1, nA2, nA3, nB;

    loadArow(arow0, 0, A0);
    loadArow(arow1, 0, A1);
    loadArow(arow2, 0, A2);
    loadArow(arow3, 0, A3);
    loadB(0, B);

    for (int kc = 0; kc < KCHUNKS; ++kc) {
        if (kc + 1 < KCHUNKS) {
            int kb = (kc + 1) * 32;
            loadArow(arow0, kb, nA0);
            loadArow(arow1, kb, nA1);
            loadArow(arow2, kb, nA2);
            loadArow(arow3, kb, nA3);
            loadB(kc + 1, nB);
        }
        acc0 = __builtin_amdgcn_wmma_f32_16x16x32_f16(false, A0, false, B, (short)0, acc0, false, false);
        acc1 = __builtin_amdgcn_wmma_f32_16x16x32_f16(false, A1, false, B, (short)0, acc1, false, false);
        acc2 = __builtin_amdgcn_wmma_f32_16x16x32_f16(false, A2, false, B, (short)0, acc2, false, false);
        acc3 = __builtin_amdgcn_wmma_f32_16x16x32_f16(false, A3, false, B, (short)0, acc3, false, false);
        A0 = nA0; A1 = nA1; A2 = nA2; A3 = nA3; B = nB;
    }

    if (!nvalid) return;

    v8f accs[4] = {acc0, acc1, acc2, acc3};
    if (out16) {
        // inner layers: O == Opad == 256, packed b128 f16 stores, fused bias+ReLU
        f16* op = out16 + (size_t)n * CIN;
        #pragma unroll
        for (int r = 0; r < 4; ++r) {
            int rb = mg * 64 + r * 16 + hi * 8;
            v8h st;
            #pragma unroll
            for (int e = 0; e < 8; ++e) {
                float v = accs[r][e] + bias[rb + e];
                if (relu) v = fmaxf(v, 0.f);
                st[e] = (f16)v;
            }
            *(v8h*)(op + rb) = st;
        }
    }
    if (out32) {
        float* op = out32 + (size_t)n * Opad;
        #pragma unroll
        for (int r = 0; r < 4; ++r) {
            int rb = mg * 64 + r * 16 + hi * 8;
            #pragma unroll
            for (int e = 0; e < 8; ++e) {
                int oc = rb + e;
                if (oc < O) {
                    float v = accs[r][e] + bias[oc];
                    if (relu) v = fmaxf(v, 0.f);
                    op[oc] = v;
                }
            }
        }
    }
}

// ---------------------------------------------------------------------------
// Per-anchor: analytic anchor box, regression decode+clip, sigmoid(max logit),
// argmax class. NHWC head layouts: cls [HW][768], reg [HW][64].
// ---------------------------------------------------------------------------
__global__ void score_decode(const float* __restrict__ clsHead,
                             const float* __restrict__ regHead,
                             int fs, int stride, float size, int anchorBase,
                             float* __restrict__ boxes,   // [NA][4]
                             float* __restrict__ masked,  // [NA]
                             int* __restrict__ clsarg)    // [NA]
{
    int HW = fs * fs;
    int nA = HW * 9;
    int idx = blockIdx.x * blockDim.x + threadIdx.x;
    if (idx >= nA) return;
    int a = idx % 9;
    int cell = idx / 9;
    int x = cell % fs;
    int y = cell / fs;

    const float ratios[3] = {0.5f, 1.0f, 2.0f};
    const float scales[3] = {1.0f, 1.2599210499f, 1.5874010520f};
    int ri = a / 3, si = a - ri * 3;
    float sr = sqrtf(ratios[ri]);
    float wa = size * scales[si] / sr;
    float ha = size * scales[si] * sr;
    float cx = (x + 0.5f) * (float)stride;
    float cy = (y + 0.5f) * (float)stride;

    const float* rp = regHead + (size_t)cell * REG_STRIDE + a * 4;
    float l0 = rp[0] * 0.1f;
    float l1 = rp[1] * 0.1f;
    float l2 = rp[2] * 0.2f;
    float l3 = rp[3] * 0.2f;
    float pcx = cx + l0 * wa;
    float pcy = cy + l1 * ha;
    float pw = expf(l2) * wa;
    float ph = expf(l3) * ha;
    float bx1 = fminf(fmaxf(pcx - 0.5f * pw, 0.f), 640.f);
    float by1 = fminf(fmaxf(pcy - 0.5f * ph, 0.f), 640.f);
    float bx2 = fminf(fmaxf(pcx + 0.5f * pw, 0.f), 640.f);
    float by2 = fminf(fmaxf(pcy + 0.5f * ph, 0.f), 640.f);

    const float* cp = clsHead + (size_t)cell * CLS_STRIDE + a * 80;
    float best = -1e30f; int bc = 0;
    for (int c = 0; c < 80; ++c) {
        float v = cp[c];
        if (v > best) { best = v; bc = c; }
    }
    float score = 1.f / (1.f + expf(-best));

    int g = anchorBase + idx;
    boxes[g * 4 + 0] = bx1; boxes[g * 4 + 1] = by1;
    boxes[g * 4 + 2] = bx2; boxes[g * 4 + 3] = by2;
    masked[g] = (score > 0.05f) ? score : -1.0f;
    clsarg[g] = bc;
}

// ---------------------------------------------------------------------------
// Exact top-1000 by iterative block-wide argmax (index tie-break = lowest,
// matching jax top_k ordering). Mutates `masked` (rewritten each launch).
// ---------------------------------------------------------------------------
__global__ void topk1000(float* __restrict__ masked, const float* __restrict__ boxes,
                         const int* __restrict__ clsarg, int N,
                         float* __restrict__ selScore, int* __restrict__ selCls,
                         float* __restrict__ selBoxes)
{
    __shared__ float sv[1024];
    __shared__ int   si[1024];
    int t = threadIdx.x;
    for (int k = 0; k < 1000; ++k) {
        float bv = -1e30f; int bi = 0;
        for (int i = t; i < N; i += 1024) {
            float v = masked[i];
            if (v > bv) { bv = v; bi = i; }   // ascending scan -> lowest idx on tie
        }
        sv[t] = bv; si[t] = bi;
        __syncthreads();
        for (int s = 512; s > 0; s >>= 1) {
            if (t < s) {
                if (sv[t + s] > sv[t] || (sv[t + s] == sv[t] && si[t + s] < si[t])) {
                    sv[t] = sv[t + s]; si[t] = si[t + s];
                }
            }
            __syncthreads();
        }
        if (t == 0) {
            int idx = si[0];
            selScore[k] = sv[0];
            selCls[k] = clsarg[idx];
            selBoxes[4 * k + 0] = boxes[4 * idx + 0];
            selBoxes[4 * k + 1] = boxes[4 * idx + 1];
            selBoxes[4 * k + 2] = boxes[4 * idx + 2];
            selBoxes[4 * k + 3] = boxes[4 * idx + 3];
            masked[idx] = -2e30f;
        }
        __syncthreads();
    }
}

// ---------------------------------------------------------------------------
// Sequential greedy NMS over the 1000 selected boxes (boxes in LDS),
// then final output packing: [scores(1000) | classes(1000) | boxes(4000) | keep(1000)]
// ---------------------------------------------------------------------------
__global__ void nms1000(const float* __restrict__ selScore, const int* __restrict__ selCls,
                        const float* __restrict__ selBoxes, float* __restrict__ out)
{
    __shared__ float bx[1000 * 4];
    __shared__ int keep[1000];
    int t = threadIdx.x;
    for (int i = t; i < 1000; i += 1024) {
        bx[4 * i + 0] = selBoxes[4 * i + 0];
        bx[4 * i + 1] = selBoxes[4 * i + 1];
        bx[4 * i + 2] = selBoxes[4 * i + 2];
        bx[4 * i + 3] = selBoxes[4 * i + 3];
        keep[i] = (selScore[i] > 0.f) ? 1 : 0;
    }
    __syncthreads();
    for (int i = 0; i < 1000; ++i) {
        if (keep[i]) {
            float x1 = bx[4 * i], y1 = bx[4 * i + 1], x2 = bx[4 * i + 2], y2 = bx[4 * i + 3];
            float ai = fmaxf(x2 - x1, 0.f) * fmaxf(y2 - y1, 0.f);
            for (int j = t; j < 1000; j += 1024) {
                if (j > i && keep[j]) {
                    float xx1 = fmaxf(x1, bx[4 * j + 0]);
                    float yy1 = fmaxf(y1, bx[4 * j + 1]);
                    float xx2 = fminf(x2, bx[4 * j + 2]);
                    float yy2 = fminf(y2, bx[4 * j + 3]);
                    float inter = fmaxf(xx2 - xx1, 0.f) * fmaxf(yy2 - yy1, 0.f);
                    float aj = fmaxf(bx[4 * j + 2] - bx[4 * j + 0], 0.f) *
                               fmaxf(bx[4 * j + 3] - bx[4 * j + 1], 0.f);
                    float iou = inter / (ai + aj - inter + 1e-8f);
                    if (iou > 0.5f) keep[j] = 0;
                }
            }
        }
        __syncthreads();
    }
    for (int i = t; i < 1000; i += 1024) {
        int kp = keep[i];
        out[i] = kp ? selScore[i] : 0.f;
        out[1000 + i] = (float)selCls[i];
        out[2000 + 4 * i + 0] = kp ? bx[4 * i + 0] : 0.f;
        out[2000 + 4 * i + 1] = kp ? bx[4 * i + 1] : 0.f;
        out[2000 + 4 * i + 2] = kp ? bx[4 * i + 2] : 0.f;
        out[2000 + 4 * i + 3] = kp ? bx[4 * i + 3] : 0.f;
        out[6000 + i] = kp ? 1.f : 0.f;
    }
}

// ---------------------------------------------------------------------------
// Host orchestration
// ---------------------------------------------------------------------------
extern "C" void kernel_launch(void* const* d_in, const int* in_sizes, int n_in,
                              void* d_out, int out_size, void* d_ws, size_t ws_size,
                              hipStream_t stream)
{
    (void)in_sizes; (void)n_in; (void)out_size; (void)ws_size;

    const float* p[5];
    for (int i = 0; i < 5; ++i) p[i] = (const float*)d_in[i];
    const float* cls_w  = (const float*)d_in[5];
    const float* cls_b  = (const float*)d_in[6];
    const float* cls_hw = (const float*)d_in[7];
    const float* cls_hb = (const float*)d_in[8];
    const float* reg_w  = (const float*)d_in[9];
    const float* reg_b  = (const float*)d_in[10];
    const float* reg_hw = (const float*)d_in[11];
    const float* reg_hb = (const float*)d_in[12];

    char* ws = (char*)d_ws;
    size_t off = 0;
    auto alloc = [&](size_t bytes) -> void* {
        off = (off + 255) & ~(size_t)255;
        void* ptr = ws + off;
        off += bytes;
        return ptr;
    };

    f16* wtCls[4]; for (int i = 0; i < 4; ++i) wtCls[i] = (f16*)alloc((size_t)256 * KTOT * 2);
    f16* wtClsH = (f16*)alloc((size_t)CLS_STRIDE * KTOT * 2);
    f16* wtReg[4]; for (int i = 0; i < 4; ++i) wtReg[i] = (f16*)alloc((size_t)256 * KTOT * 2);
    f16* wtRegH = (f16*)alloc((size_t)REG_STRIDE * KTOT * 2);
    f16* feat = (f16*)alloc((size_t)6400 * CIN * 2);
    f16* tA   = (f16*)alloc((size_t)6400 * CIN * 2);
    f16* tB   = (f16*)alloc((size_t)6400 * CIN * 2);
    float* clsHead = (float*)alloc((size_t)6400 * CLS_STRIDE * 4);
    float* regHead = (float*)alloc((size_t)6400 * REG_STRIDE * 4);
    float* boxes   = (float*)alloc((size_t)76725 * 4 * 4);
    float* masked  = (float*)alloc((size_t)76725 * 4);
    int*   clsarg  = (int*)alloc((size_t)76725 * 4);
    float* selScore = (float*)alloc(1000 * 4);
    int*   selCls   = (int*)alloc(1000 * 4);
    float* selBoxes = (float*)alloc(4000 * 4);

    // weight repacks (once per launch; deterministic)
    for (int i = 0; i < 4; ++i) {
        int n = 256 * KTOT;
        wt_transform<<<(n + 255) / 256, 256, 0, stream>>>(cls_w + (size_t)i * 256 * 256 * 9, wtCls[i], 256, 256);
        wt_transform<<<(n + 255) / 256, 256, 0, stream>>>(reg_w + (size_t)i * 256 * 256 * 9, wtReg[i], 256, 256);
    }
    { int n = CLS_STRIDE * KTOT; wt_transform<<<(n + 255) / 256, 256, 0, stream>>>(cls_hw, wtClsH, 720, CLS_STRIDE); }
    { int n = REG_STRIDE * KTOT; wt_transform<<<(n + 255) / 256, 256, 0, stream>>>(reg_hw, wtRegH, 36, REG_STRIDE); }

    auto conv = [&](const f16* in, const f16* wt, const float* bias,
                    f16* o16, float* o32, int H, int O, int Opad, int relu) {
        int HW = H * H;
        int nTilesN = (HW + 15) / 16;
        int T = (Opad / 64) * nTilesN;
        conv3x3_wmma<<<(T + 3) / 4, 128, 0, stream>>>(in, wt, bias, o16, o32,
                                                      H, H, HW, O, Opad, nTilesN, T, relu);
    };

    const int   HS[5]  = {80, 40, 20, 10, 5};
    const int   STR[5] = {8, 16, 32, 64, 128};
    const float SZ[5]  = {32.f, 64.f, 128.f, 256.f, 512.f};
    int anchorBase = 0;
    for (int l = 0; l < 5; ++l) {
        int H = HS[l], HW = H * H;
        int n = CIN * HW;
        cvt_nchw_to_nhwc_f16<<<(n + 255) / 256, 256, 0, stream>>>(p[l], feat, HW);
        // classification head
        conv(feat, wtCls[0], cls_b + 0,   tA, nullptr, H, 256, 256, 1);
        conv(tA,   wtCls[1], cls_b + 256, tB, nullptr, H, 256, 256, 1);
        conv(tB,   wtCls[2], cls_b + 512, tA, nullptr, H, 256, 256, 1);
        conv(tA,   wtCls[3], cls_b + 768, tB, nullptr, H, 256, 256, 1);
        conv(tB,   wtClsH,   cls_hb, nullptr, clsHead, H, 720, CLS_STRIDE, 0);
        // regression head
        conv(feat, wtReg[0], reg_b + 0,   tA, nullptr, H, 256, 256, 1);
        conv(tA,   wtReg[1], reg_b + 256, tB, nullptr, H, 256, 256, 1);
        conv(tB,   wtReg[2], reg_b + 512, tA, nullptr, H, 256, 256, 1);
        conv(tA,   wtReg[3], reg_b + 768, tB, nullptr, H, 256, 256, 1);
        conv(tB,   wtRegH,   reg_hb, nullptr, regHead, H, 36, REG_STRIDE, 0);
        // decode + score this level's anchors
        int nA = HW * 9;
        score_decode<<<(nA + 255) / 256, 256, 0, stream>>>(clsHead, regHead, H, STR[l], SZ[l],
                                                           anchorBase, boxes, masked, clsarg);
        anchorBase += nA;
    }

    topk1000<<<1, 1024, 0, stream>>>(masked, boxes, clsarg, 76725, selScore, selCls, selBoxes);
    nms1000<<<1, 1024, 0, stream>>>(selScore, selCls, selBoxes, (float*)d_out);
}
